// GambaAR_6030134083934
// MI455X (gfx1250) — matrix operand order repaired and checked by reading, not compiled, not run on hardware
//
#include <hip/hip_runtime.h>
#include <hip/hip_bf16.h>
#include <math.h>

// ---------------------------------------------------------------------------
// GambaAR pipeline for MI455X (gfx1250, wave32, WMMA).
// All GEMMs go through v_wmma_f32_16x16x32_f16 with f16 LDS tiles, f32 acc.
// ---------------------------------------------------------------------------

typedef __attribute__((ext_vector_type(16))) _Float16 v16h;
typedef __attribute__((ext_vector_type(8)))  _Float16 v8h;
typedef __attribute__((ext_vector_type(4)))  _Float16 v4h;
typedef __attribute__((ext_vector_type(8)))  float    v8f;

static constexpr int NN   = 50000;   // nodes
static constexpr long EE  = 800000;  // edges
static constexpr int BG   = 100;     // graphs
static constexpr int MNOD = 500;     // nodes per graph
static constexpr int DM   = 128;     // 2H (mamba d_model)
static constexpr int DI   = 256;     // d_inner
static constexpr int DS   = 128;     // d_state
static constexpr int DTR  = 8;       // dt_rank
static constexpr int LSEQ = 4;       // virtual tokens

// ===========================================================================
// Generic WMMA GEMM:  C[M,N] = A[M,K] @ W[K,N] (+ bias[N])
// Block = 128 threads (4 waves). Block tile = 32 rows x 64 cols.
// Each wave owns one 16-col strip and two 16x16 row tiles (B-fragment reuse).
// Interior tiles: vectorized b128 global loads + b64/b128 LDS ops.
// Edge tiles (M%32, K%32, N%64 tails): guarded scalar path (block-uniform
// selection, so EXEC is all-1s at every v_wmma).
// ===========================================================================
__launch_bounds__(128)
__global__ void wmma_gemm_kernel(const float* __restrict__ A, int lda,
                                 const float* __restrict__ W, int ldw,
                                 const float* __restrict__ bias,
                                 float* __restrict__ C, int ldc,
                                 int M, int N, int K) {
  __shared__ alignas(16) _Float16 sA[32 * 32];   // [row][k]   row-major
  __shared__ alignas(16) _Float16 sBt[64 * 32];  // [col][k]   transposed
  const int tid  = threadIdx.x;
  const int lane = tid & 31;
  const int wave = tid >> 5;
  const int rowBase = blockIdx.x * 32;
  const int colBase = blockIdx.y * 64;
  const bool interior = (rowBase + 32 <= M) && (colBase + 64 <= N);

  v8f acc0 = {}, acc1 = {};
  for (int k0 = 0; k0 < K; k0 += 32) {
    if (interior && (k0 + 32 <= K)) {
      // ---- fast path: vectorized stage ----
      {
        // A: 32x32 tile, each thread loads 8 contiguous floats of one row
        const int r  = tid >> 2;          // 0..31
        const int c0 = (tid & 3) << 3;    // 0,8,16,24
        const float* pa = A + (size_t)(rowBase + r) * lda + k0 + c0;
        const float4 va = *(const float4*)pa;
        const float4 vb = *(const float4*)(pa + 4);
        v8h h;
        h[0] = (_Float16)va.x; h[1] = (_Float16)va.y;
        h[2] = (_Float16)va.z; h[3] = (_Float16)va.w;
        h[4] = (_Float16)vb.x; h[5] = (_Float16)vb.y;
        h[6] = (_Float16)vb.z; h[7] = (_Float16)vb.w;
        *(v8h*)&sA[r * 32 + c0] = h;
        if (k0 + 32 < K) __builtin_prefetch(pa + 32, 0, 1);   // next K slab
      }
      {
        // B: 32x64 tile as 4x4 sub-block transpose per thread
        const int br = (tid >> 4) << 2;   // 0..28 step 4  (k rows)
        const int bc = (tid & 15) << 2;   // 0..60 step 4  (cols)
        float4 rv[4];
#pragma unroll
        for (int i = 0; i < 4; ++i)
          rv[i] = *(const float4*)(W + (size_t)(k0 + br + i) * ldw + colBase + bc);
#pragma unroll
        for (int j = 0; j < 4; ++j) {
          v4h h;
          h[0] = (_Float16)(&rv[0].x)[j];
          h[1] = (_Float16)(&rv[1].x)[j];
          h[2] = (_Float16)(&rv[2].x)[j];
          h[3] = (_Float16)(&rv[3].x)[j];
          *(v4h*)&sBt[(bc + j) * 32 + br] = h;
        }
        if (k0 + 32 < K)
          __builtin_prefetch(W + (size_t)(k0 + 32 + br) * ldw + colBase + bc, 0, 1);
      }
    } else {
      // ---- edge path: guarded scalar stage (zero padded) ----
      for (int idx = tid; idx < 32 * 32; idx += 128) {
        int r = idx >> 5, c = idx & 31;
        int gr = rowBase + r, gc = k0 + c;
        float v = (gr < M && gc < K) ? A[(size_t)gr * lda + gc] : 0.0f;
        sA[r * 32 + c] = (_Float16)v;
      }
      for (int idx = tid; idx < 32 * 64; idx += 128) {
        int k = idx >> 6, c = idx & 63;
        int gk = k0 + k, gc = colBase + c;
        float v = (gk < K && gc < N) ? W[(size_t)gk * ldw + gc] : 0.0f;
        sBt[c * 32 + k] = (_Float16)v;
      }
    }
    __syncthreads();

    // ---- fragments (CDNA5 ISA 7.12.2 wave32 layouts), b128 LDS loads ----
    union { v16h v; v8h h[2]; } af0, af1, bf;
    {
      const int r  = lane & 15;          // A: M = lane%16
      const int kb = (lane >> 4) * 8;    // lanes 16-31: +8 K offset
      af0.h[0] = *(const v8h*)&sA[r * 32 + kb];            // K kb..kb+7
      af0.h[1] = *(const v8h*)&sA[r * 32 + kb + 16];       // K kb+16..kb+23
      af1.h[0] = *(const v8h*)&sA[(16 + r) * 32 + kb];
      af1.h[1] = *(const v8h*)&sA[(16 + r) * 32 + kb + 16];
      const int cc  = wave * 16 + (lane & 15);             // B: N = lane%16
      const int kb2 = (lane >> 4) * 8;                     // lanes 16-31: K+8
      bf.h[0] = *(const v8h*)&sBt[cc * 32 + kb2];
      bf.h[1] = *(const v8h*)&sBt[cc * 32 + kb2 + 16];
    }
    acc0 = __builtin_amdgcn_wmma_f32_16x16x32_f16(false, af0.v, false, bf.v,
                                                  (short)0, acc0, false, false);
    acc1 = __builtin_amdgcn_wmma_f32_16x16x32_f16(false, af1.v, false, bf.v,
                                                  (short)0, acc1, false, false);
    __syncthreads();
  }

  // store: D layout — VGPR v: M = v + 8*(lane>=16); N = lane%16
  const int cg = colBase + wave * 16 + (lane & 15);
  const float bv = (bias && cg < N) ? bias[cg] : 0.0f;
#pragma unroll
  for (int vv = 0; vv < 8; ++vv) {
    int r0 = rowBase + vv + ((lane >> 4) * 8);
    if (r0 < M && cg < N)      C[(size_t)r0 * ldc + cg]        = acc0[vv] + bv;
    if (r0 + 16 < M && cg < N) C[(size_t)(r0 + 16) * ldc + cg] = acc1[vv] + bv;
  }
}

// ===========================================================================
// Row-wise LayerNorm (+ optional ReLU), in place. One block (64 thr) per row.
// ===========================================================================
__global__ void ln_act_kernel(float* __restrict__ X, int Mrows, int D,
                              const float* __restrict__ g,
                              const float* __restrict__ bta, int relu) {
  int row = blockIdx.x;
  if (row >= Mrows) return;
  float* xr = X + (size_t)row * D;
  __shared__ float s1[64], s2[64];
  int t = threadIdx.x;
  float a = 0.f, b2 = 0.f;
  for (int c = t; c < D; c += 64) { float v = xr[c]; a += v; b2 += v * v; }
  s1[t] = a; s2[t] = b2; __syncthreads();
  for (int off = 32; off > 0; off >>= 1) {
    if (t < off) { s1[t] += s1[t + off]; s2[t] += s2[t + off]; }
    __syncthreads();
  }
  float mean = s1[0] / D;
  float var  = s2[0] / D - mean * mean;
  float inv  = rsqrtf(var + 1e-5f);
  for (int c = t; c < D; c += 64) {
    float v = (xr[c] - mean) * inv * g[c] + bta[c];
    if (relu) v = fmaxf(v, 0.f);
    xr[c] = v;
  }
}

// ===========================================================================
// Edge scatter-add: out[dst[e], c] += feat[src[e], c]   (segment_sum)
// ===========================================================================
__global__ void scatter_add_kernel(const float* __restrict__ feat,
                                   const int* __restrict__ ei,
                                   float* __restrict__ out, long E, int C) {
  long total = E * (long)C;
  for (long i = (long)blockIdx.x * blockDim.x + threadIdx.x; i < total;
       i += (long)gridDim.x * blockDim.x) {
    long e = i / C;
    int  c = (int)(i - e * C);
    int srcn = ei[e];
    int dstn = ei[E + e];
    atomicAdd(&out[(size_t)dstn * C + c], feat[(size_t)srcn * C + c]);
  }
}

// ===========================================================================
// GRU gate combine (elementwise): h' = (1-z)*n + z*h
// ===========================================================================
__global__ void gru_gate_kernel(const float* __restrict__ gi,
                                const float* __restrict__ gh,
                                const float* __restrict__ hprev,
                                float* __restrict__ hout, int Mrows, int Hdim) {
  long i = (long)blockIdx.x * blockDim.x + threadIdx.x;
  long total = (long)Mrows * Hdim;
  if (i >= total) return;
  int m = (int)(i / Hdim), c = (int)(i % Hdim);
  const float* gir = gi + (size_t)m * 3 * Hdim;
  const float* ghr = gh + (size_t)m * 3 * Hdim;
  float r = 1.f / (1.f + expf(-(gir[c] + ghr[c])));
  float z = 1.f / (1.f + expf(-(gir[Hdim + c] + ghr[Hdim + c])));
  float n = tanhf(gir[2 * Hdim + c] + r * ghr[2 * Hdim + c]);
  float h = hprev[(size_t)m * Hdim + c];
  hout[(size_t)m * Hdim + c] = (1.f - z) * n + z * h;
}

// inp[n] = [ xenc[n](64) | pe[n](64) | h[graph(n)](64) ]
__global__ void build_inp_kernel(const float* __restrict__ xenc,
                                 const float* __restrict__ pe,
                                 const float* __restrict__ h,
                                 float* __restrict__ inp) {
  int i = blockIdx.x * blockDim.x + threadIdx.x;
  if (i >= NN * 192) return;
  int n = i / 192, c = i % 192;
  float v;
  if (c < 64)       v = xenc[n * 64 + c];
  else if (c < 128) v = pe[n * 64 + (c - 64)];
  else              v = h[(n / MNOD) * 64 + (c - 128)];
  inp[i] = v;
}

// minp[n] = [ xenc[n](64) | xm_last[graph(n)](128) ]
__global__ void build_minp_kernel(const float* __restrict__ xenc,
                                  const float* __restrict__ xm,
                                  float* __restrict__ minp) {
  int i = blockIdx.x * blockDim.x + threadIdx.x;
  if (i >= NN * 192) return;
  int n = i / 192, c = i % 192;
  float v;
  if (c < 64) v = xenc[n * 64 + c];
  else        v = xm[((n / MNOD) * LSEQ + (LSEQ - 1)) * DM + (c - 64)];
  minp[i] = v;
}

// ===========================================================================
// Attention pooling: t[b,j] = sum_m alpha[b,m,j] * xcat[b,m,j]; xcat=[xenc|pe]
// ===========================================================================
__global__ void attn_pool_kernel(const float* __restrict__ alpha,
                                 const float* __restrict__ xenc,
                                 const float* __restrict__ pe,
                                 float* __restrict__ ts, int it) {
  int b = blockIdx.x;
  int j = threadIdx.x;   // 0..127
  float acc = 0.f;
  for (int m = 0; m < MNOD; ++m) {
    int n = b * MNOD + m;
    float xc = (j < 64) ? xenc[(size_t)n * 64 + j] : pe[(size_t)n * 64 + (j - 64)];
    acc += alpha[(size_t)n * DM + j] * xc;
  }
  ts[(b * LSEQ + it) * DM + j] = acc;
}

// ===========================================================================
// Mamba: causal depthwise conv (D_CONV=4) + SiLU. One thread per (b, ch).
// ===========================================================================
__global__ void conv_silu_kernel(const float* __restrict__ xz,
                                 const float* __restrict__ cw,
                                 const float* __restrict__ cb,
                                 float* __restrict__ xs) {
  int i = blockIdx.x * blockDim.x + threadIdx.x;
  if (i >= BG * DI) return;
  int b = i / DI, ch = i % DI;
  float xv[LSEQ];
#pragma unroll
  for (int l = 0; l < LSEQ; ++l) xv[l] = xz[(size_t)(b * LSEQ + l) * (2 * DI) + ch];
#pragma unroll
  for (int l = 0; l < LSEQ; ++l) {
    float acc = cb[ch];
#pragma unroll
    for (int k = 0; k < 4; ++k) {
      int t = l - 3 + k;
      if (t >= 0) acc += cw[ch * 4 + k] * xv[t];
    }
    float s = acc / (1.f + expf(-acc));       // SiLU
    xs[(size_t)(b * LSEQ + l) * DI + ch] = s;
  }
}

__global__ void softplus_kernel(float* __restrict__ x, int n) {
  int i = blockIdx.x * blockDim.x + threadIdx.x;
  if (i >= n) return;
  float v = x[i];
  x[i] = (v > 20.f) ? v : log1pf(expf(v));
}

// ===========================================================================
// Selective-scan: block = one (b, ch), 128 threads = state dim.
// h_s = exp(dt*A_s)*h_s + dt*B_s*x ; y = sum_s h_s*C_s + x*D ; ym = y*silu(z)
// ===========================================================================
__global__ void ssm_scan_kernel(const float* __restrict__ dt,
                                const float* __restrict__ xs,
                                const float* __restrict__ dtBC,
                                const float* __restrict__ xz,
                                const float* __restrict__ A_log,
                                const float* __restrict__ Dp,
                                float* __restrict__ ym) {
  int bi = blockIdx.x;           // 0 .. BG*DI-1
  int b  = bi / DI, ch = bi % DI;
  int s  = threadIdx.x;          // 0..127
  __shared__ float red[128];
  float Aval = -expf(A_log[ch * DS + s]);
  float h = 0.f;
  for (int l = 0; l < LSEQ; ++l) {
    int row = b * LSEQ + l;
    float dtv = dt[(size_t)row * DI + ch];
    float xv  = xs[(size_t)row * DI + ch];
    float Bv  = dtBC[(size_t)row * (DTR + 2 * DS) + DTR + s];
    float Cv  = dtBC[(size_t)row * (DTR + 2 * DS) + DTR + DS + s];
    h = expf(dtv * Aval) * h + dtv * Bv * xv;
    red[s] = h * Cv;
    __syncthreads();
    for (int off = 64; off > 0; off >>= 1) {
      if (s < off) red[s] += red[s + off];
      __syncthreads();
    }
    if (s == 0) {
      float y  = red[0] + xv * Dp[ch];
      float zv = xz[(size_t)row * (2 * DI) + DI + ch];
      ym[(size_t)row * DI + ch] = y * (zv / (1.f + expf(-zv)));
    }
    __syncthreads();
  }
}

// global_add_pool: pooled[b,c] = sum_m xg[b*500+m, c]
__global__ void pool_sum_kernel(const float* __restrict__ xg,
                                float* __restrict__ pooled) {
  int b = blockIdx.x;
  int c = threadIdx.x;  // 64
  float acc = 0.f;
  for (int m = 0; m < MNOD; ++m) acc += xg[(size_t)(b * MNOD + m) * 64 + c];
  pooled[b * 64 + c] = acc;
}

// ===========================================================================
// Host orchestration
// ===========================================================================
static inline void gemm(const float* A, int lda, const float* W, int ldw,
                        const float* bias, float* C, int ldc,
                        int M, int N, int K, hipStream_t stream) {
  dim3 grid((M + 31) / 32, (N + 63) / 64);
  hipLaunchKernelGGL(wmma_gemm_kernel, grid, dim3(128), 0, stream,
                     A, lda, W, ldw, bias, C, ldc, M, N, K);
}

extern "C" void kernel_launch(void* const* d_in, const int* in_sizes, int n_in,
                              void* d_out, int out_size, void* d_ws, size_t ws_size,
                              hipStream_t stream) {
  // --- inputs (top-level insertion order; params pytree key-sorted) ---
  const float* x   = (const float*)d_in[0];            // [N,64]
  const int*   ei  = (const int*)d_in[1];              // [2,E]
  // d_in[2] = batch (unused: batch[n] == n/500)
  const float* dec0W = (const float*)d_in[3];
  const float* dec0b = (const float*)d_in[4];
  const float* dec0bt= (const float*)d_in[5];
  const float* dec0g = (const float*)d_in[6];
  const float* dec1W = (const float*)d_in[7];
  const float* dec1b = (const float*)d_in[8];
  const float* enc0W = (const float*)d_in[9];
  const float* enc0b = (const float*)d_in[10];
  const float* enc0bt= (const float*)d_in[11];
  const float* enc0g = (const float*)d_in[12];
  const float* enc1W = (const float*)d_in[13];
  const float* enc1b = (const float*)d_in[14];
  const float* ggW   = (const float*)d_in[15];         // [8,64,64]
  const float* ggWhh = (const float*)d_in[16];         // [64,192]
  const float* ggWih = (const float*)d_in[17];         // [64,192]
  const float* ggbhh = (const float*)d_in[18];
  const float* ggbih = (const float*)d_in[19];
  const float* ginW  = (const float*)d_in[20];
  const float* ginb  = (const float*)d_in[21];
  const float* grWhh = (const float*)d_in[22];         // [64,192]
  const float* grWih = (const float*)d_in[23];         // [128,192]
  const float* grbhh = (const float*)d_in[24];
  const float* grbih = (const float*)d_in[25];
  const float* ln_b  = (const float*)d_in[26];
  const float* ln_g  = (const float*)d_in[27];
  const float* mA    = (const float*)d_in[28];         // A_log [256,128]
  const float* mD    = (const float*)d_in[29];         // [256]
  const float* mWdt  = (const float*)d_in[30];         // [8,256]
  const float* mWin  = (const float*)d_in[31];         // [128,512]
  const float* mWout = (const float*)d_in[32];         // [256,128]
  const float* mWx   = (const float*)d_in[33];         // [256,264]
  const float* mbdt  = (const float*)d_in[34];
  const float* mcb   = (const float*)d_in[35];
  const float* mcw   = (const float*)d_in[36];         // [256,4]
  const float* mrgW  = (const float*)d_in[37];         // [192,64]
  const float* mrgb  = (const float*)d_in[38];
  const float* w0W   = (const float*)d_in[39];         // [192,192]
  const float* w0b   = (const float*)d_in[40];
  const float* w0bt  = (const float*)d_in[41];
  const float* w0g   = (const float*)d_in[42];
  const float* w1W   = (const float*)d_in[43];         // [192,128]
  const float* w1b   = (const float*)d_in[44];

  // --- workspace layout (floats) ---
  float* ws = (float*)d_ws;
  const size_t NH = (size_t)NN * 64, N192 = (size_t)NN * 192, N128 = (size_t)NN * 128;
  float* xenc  = ws;                 // x_orig [N,64]
  float* pe    = xenc + NH;          // [N,64]
  float* msg   = pe + NH;            // [N,64]  (later xf)
  float* agg   = msg + NH;           // [N,64]  (later xf+seg)
  float* gi    = agg + NH;           // [N,192] (later inp / minp)
  float* gh    = gi + N192;          // [N,192] (later a1 / xg)
  float* alpha = gh + N192;          // [N,128]
  float* sb    = alpha + N128;       // small buffers
  float* ts    = sb;                       sb += BG * LSEQ * DM;     // [400,128]
  float* hbuf  = sb;                       sb += BG * 64;
  float* gis   = sb;                       sb += BG * 192;
  float* ghs   = sb;                       sb += BG * 192;
  float* xzb   = sb;                       sb += BG * LSEQ * 2 * DI; // [400,512]
  float* xsb   = sb;                       sb += BG * LSEQ * DI;     // [400,256]
  float* dtBC  = sb;                       sb += BG * LSEQ * (DTR + 2 * DS); // [400,264]
  float* dtb   = sb;                       sb += BG * LSEQ * DI;
  float* ymb   = sb;                       sb += BG * LSEQ * DI;
  float* xmb   = sb;                       sb += BG * LSEQ * DM;     // [400,128]
  float* pooled= sb;                       sb += BG * 64;
  float* dech  = sb;                       sb += BG * 64;
  (void)ws_size; (void)in_sizes; (void)n_in; (void)out_size;

  const int EB = 2048, ET = 256;  // scatter grid
  // ---------------- encoder MLP ----------------
  gemm(x, 64, enc0W, 64, enc0b, msg, 64, NN, 64, 64, stream);
  hipLaunchKernelGGL(ln_act_kernel, dim3(NN), dim3(64), 0, stream, msg, NN, 64, enc0g, enc0bt, 1);
  gemm(msg, 64, enc1W, 64, enc1b, xenc, 64, NN, 64, 64, stream);
  hipMemcpyAsync(pe, xenc, NH * sizeof(float), hipMemcpyDeviceToDevice, stream);

  // ---------------- GatedGraphConv x8 ----------------
  for (int layer = 0; layer < 8; ++layer) {
    gemm(pe, 64, ggW + (size_t)layer * 64 * 64, 64, nullptr, msg, 64, NN, 64, 64, stream);
    hipMemsetAsync(agg, 0, NH * sizeof(float), stream);
    hipLaunchKernelGGL(scatter_add_kernel, dim3(EB), dim3(ET), 0, stream, msg, ei, agg, EE, 64);
    gemm(agg, 64, ggWih, 192, ggbih, gi, 192, NN, 192, 64, stream);
    gemm(pe, 64, ggWhh, 192, ggbhh, gh, 192, NN, 192, 64, stream);
    hipLaunchKernelGGL(gru_gate_kernel, dim3((NN * 64 + 255) / 256), dim3(256), 0, stream,
                       gi, gh, pe, pe, NN, 64);
  }

  // ---------------- virtual-token loop ----------------
  hipMemsetAsync(hbuf, 0, BG * 64 * sizeof(float), stream);
  for (int it = 0; it < LSEQ; ++it) {
    hipLaunchKernelGGL(build_inp_kernel, dim3((NN * 192 + 255) / 256), dim3(256), 0, stream,
                       xenc, pe, hbuf, gi);
    gemm(gi, 192, w0W, 192, w0b, gh, 192, NN, 192, 192, stream);
    hipLaunchKernelGGL(ln_act_kernel, dim3(NN), dim3(64), 0, stream, gh, NN, 192, w0g, w0bt, 1);
    gemm(gh, 192, w1W, 128, w1b, alpha, 128, NN, 128, 192, stream);
    hipLaunchKernelGGL(attn_pool_kernel, dim3(BG), dim3(128), 0, stream, alpha, xenc, pe, ts, it);
    // GRUCell on [B=100]: gi = t @ Wih + bih ; gh = h @ Whh + bhh
    gemm(ts + it * DM, LSEQ * DM, grWih, 192, grbih, gis, 192, BG, 192, 128, stream);
    gemm(hbuf, 64, grWhh, 192, grbhh, ghs, 192, BG, 192, 64, stream);
    hipLaunchKernelGGL(gru_gate_kernel, dim3((BG * 64 + 255) / 256), dim3(256), 0, stream,
                       gis, ghs, hbuf, hbuf, BG, 64);
  }

  // ---------------- Mamba over [B,4,128] ----------------
  gemm(ts, DM, mWin, 2 * DI, nullptr, xzb, 2 * DI, BG * LSEQ, 2 * DI, DM, stream);
  hipLaunchKernelGGL(conv_silu_kernel, dim3((BG * DI + 255) / 256), dim3(256), 0, stream,
                     xzb, mcw, mcb, xsb);
  gemm(xsb, DI, mWx, DTR + 2 * DS, nullptr, dtBC, DTR + 2 * DS, BG * LSEQ, DTR + 2 * DS, DI, stream);
  gemm(dtBC, DTR + 2 * DS, mWdt, DI, mbdt, dtb, DI, BG * LSEQ, DI, DTR, stream);
  hipLaunchKernelGGL(softplus_kernel, dim3((BG * LSEQ * DI + 255) / 256), dim3(256), 0, stream,
                     dtb, BG * LSEQ * DI);
  hipLaunchKernelGGL(ssm_scan_kernel, dim3(BG * DI), dim3(DS), 0, stream,
                     dtb, xsb, dtBC, xzb, mA, mD, ymb);
  gemm(ymb, DI, mWout, DM, nullptr, xmb, DM, BG * LSEQ, DM, DI, stream);
  hipLaunchKernelGGL(ln_act_kernel, dim3(BG * LSEQ), dim3(64), 0, stream, xmb, BG * LSEQ, DM, ln_g, ln_b, 0);

  // ---------------- merge + GIN + pool + decoder ----------------
  hipLaunchKernelGGL(build_minp_kernel, dim3((NN * 192 + 255) / 256), dim3(256), 0, stream,
                     xenc, xmb, gi);
  gemm(gi, 192, mrgW, 64, mrgb, msg, 64, NN, 64, 192, stream);          // xf
  hipMemcpyAsync(agg, msg, NH * sizeof(float), hipMemcpyDeviceToDevice, stream);
  hipLaunchKernelGGL(scatter_add_kernel, dim3(EB), dim3(ET), 0, stream, msg, ei, agg, EE, 64);
  gemm(agg, 64, ginW, 64, ginb, gh, 64, NN, 64, 64, stream);            // xg -> gh region
  hipLaunchKernelGGL(pool_sum_kernel, dim3(BG), dim3(64), 0, stream, gh, pooled);
  gemm(pooled, 64, dec0W, 64, dec0b, dech, 64, BG, 64, 64, stream);
  hipLaunchKernelGGL(ln_act_kernel, dim3(BG), dim3(64), 0, stream, dech, BG, 64, dec0g, dec0bt, 1);
  gemm(dech, 64, dec1W, 64, dec1b, (float*)d_out, 64, BG, 64, 64, stream);
}
